// Model_13846974562554
// MI455X (gfx1250) — compile-verified, matrix operand order checked by reading, not run on hardware
//
#include <hip/hip_runtime.h>
#include <cstdint>
#include <cstddef>

// ---------------------------------------------------------------------------
// GPT-2-like forward for MI455X (gfx1250, wave32, WMMA + async LDS copies).
// All matmuls use V_WMMA_F32_16X16X32_F16 (f16 A/B, f32 accumulate).
// Weights are pre-converted to f16 and transposed to [N,K] (zero-padded N)
// so GEMM tiles are pure 16B copies done with GLOBAL_LOAD_ASYNC_TO_LDS_B128.
// ---------------------------------------------------------------------------

typedef _Float16 v16h __attribute__((ext_vector_type(16)));
typedef float    v8f  __attribute__((ext_vector_type(8)));

#define WMMA_F32_F16(a, b, c) \
  __builtin_amdgcn_wmma_f32_16x16x32_f16(false, (a), false, (b), (short)0, (c), false, false)

constexpr int Bc  = 2;
constexpr int Sc  = 1024;
constexpr int Dc  = 768;
constexpr int Hc  = 12;
constexpr int Lc  = 12;
constexpr int Vc  = 50257;
constexpr int DHc = 64;
constexpr int HIDc = 9216;
constexpr int Rc  = Bc * Sc;            // 2048 token rows
constexpr int D3  = 3 * Dc;             // 2304
constexpr int Vpad = ((Vc + 127) / 128) * 128;   // 50304

// Async copy of one 16-byte chunk from global to LDS (CDNA5, ASYNCcnt).
__device__ inline void async_cp16(void* lds, const _Float16* g) {
  unsigned l = (unsigned)(size_t)lds;                 // low 32 bits = LDS offset
  unsigned long long ga = (unsigned long long)(size_t)g;
  asm volatile("global_load_async_to_lds_b128 %0, %1, off"
               :: "v"(l), "v"(ga) : "memory");
}
__device__ inline void wait_async0() {
  asm volatile("s_wait_asynccnt 0x0" ::: "memory");
}

// Load a 16-bit A/B WMMA fragment from a row-major row pointer.
//   g = 8*(lane>=16); elements 0..7 -> K = g..g+7, 8..15 -> K = 16+g..23+g.
__device__ inline v16h ld_frag(const _Float16* rowp, int g) {
  v16h f;
#pragma unroll
  for (int e = 0; e < 8; ++e) {
    f[e]     = rowp[g + e];
    f[e + 8] = rowp[16 + g + e];
  }
  return f;
}

// ---------------------------------------------------------------------------
// Embedding: x[r,d] = vocab_W[vocab[r], d] + pos_W[pos[r], d]
// ---------------------------------------------------------------------------
__global__ void embed_kernel(const int* __restrict__ vocab, const int* __restrict__ pos,
                             const float* __restrict__ vW, const float* __restrict__ pW,
                             float* __restrict__ x) {
  int idx = blockIdx.x * blockDim.x + threadIdx.x;
  if (idx >= Rc * Dc) return;
  int r = idx / Dc, d = idx % Dc;
  x[idx] = vW[(size_t)vocab[r] * Dc + d] + pW[(size_t)pos[r] * Dc + d];
}

// ---------------------------------------------------------------------------
// Weight transpose+convert: W[K,N] f32 -> Wt[Npad,K] f16 (rows >= N zeroed).
// Grid: (Npad/32, K/32), block (32,8).
// ---------------------------------------------------------------------------
__global__ __launch_bounds__(256) void wt_kernel(const float* __restrict__ W,
                                                 _Float16* __restrict__ Wt,
                                                 int K, int N) {
  __shared__ float sh[32][33];
  int tx = threadIdx.x, ty = threadIdx.y;
  int n0 = blockIdx.x * 32, k0 = blockIdx.y * 32;
#pragma unroll
  for (int r = 0; r < 32; r += 8) {
    int k = k0 + ty + r, n = n0 + tx;
    sh[ty + r][tx] = (n < N) ? W[(size_t)k * N + n] : 0.0f;
  }
  __syncthreads();
#pragma unroll
  for (int r = 0; r < 32; r += 8) {
    int n = n0 + ty + r, k = k0 + tx;
    Wt[(size_t)n * K + k] = (_Float16)sh[tx][ty + r];
  }
}

// ---------------------------------------------------------------------------
// LayerNorm over D=768 (one block per row). Writes f32 and/or f16 outputs.
// ---------------------------------------------------------------------------
__global__ __launch_bounds__(256) void ln_kernel(const float* __restrict__ in,
                                                 const float* __restrict__ gamma,
                                                 const float* __restrict__ beta,
                                                 float* __restrict__ outf,
                                                 _Float16* __restrict__ outh) {
  __shared__ float red[256], red2[256];
  int r = blockIdx.x, t = threadIdx.x;
  const float* row = in + (size_t)r * Dc;
  float v0 = row[t], v1 = row[t + 256], v2 = row[t + 512];
  red[t]  = v0 + v1 + v2;
  red2[t] = v0 * v0 + v1 * v1 + v2 * v2;
  __syncthreads();
  for (int st = 128; st > 0; st >>= 1) {
    if (t < st) { red[t] += red[t + st]; red2[t] += red2[t + st]; }
    __syncthreads();
  }
  float mean = red[0] * (1.0f / Dc);
  float var  = red2[0] * (1.0f / Dc) - mean * mean;
  float inv  = rsqrtf(var + 1e-5f);
#pragma unroll
  for (int j = 0; j < 3; ++j) {
    int d = t + j * 256;
    float v = (j == 0 ? v0 : (j == 1 ? v1 : v2));
    float y = (v - mean) * inv * gamma[d] + beta[d];
    if (outf) outf[(size_t)r * Dc + d] = y;
    if (outh) outh[(size_t)r * Dc + d] = (_Float16)y;
  }
}

// ---------------------------------------------------------------------------
// WMMA GEMM: C[M,N] = A[M,K](f16, row-major) * Bt[Npad,K](f16, N-major)
//   (+bias)(+leaky_relu)(+resid). Block: 256 thr = 8 waves; tile 64x128;
//   each wave owns a 32x32 C tile (4 WMMAs/K-step, 2x fragment reuse).
//   Tiles fetched with double-buffered async global->LDS b128 copies.
// ---------------------------------------------------------------------------
__global__ __launch_bounds__(256) void gemm_kernel(const _Float16* __restrict__ A,
                                                   const _Float16* __restrict__ Bt,
                                                   const float* __restrict__ bias,
                                                   const float* __restrict__ resid,
                                                   float* __restrict__ outf,
                                                   _Float16* __restrict__ outh,
                                                   int N, int K, int act) {
  __shared__ alignas(16) _Float16 As[2][64][40];
  __shared__ alignas(16) _Float16 Bs[2][128][40];
  int tid  = threadIdx.x;
  int lane = tid & 31, ln = lane & 15, g = (lane >> 4) * 8;
  int w = tid >> 5, wr = w >> 2, wc = w & 3;        // 2x4 wave grid
  int bm = blockIdx.y * 64, bn = blockIdx.x * 128;

  v8f acc[4];
#pragma unroll
  for (int q = 0; q < 4; ++q)
#pragma unroll
    for (int i = 0; i < 8; ++i) acc[q][i] = 0.0f;

  // copy coordinates: 16B chunk per (row, chunk) pair
  int cr = tid >> 2, cc = (tid & 3) * 8;            // cr in 0..63, cc in {0,8,16,24}
  const _Float16* Ab  = A  + (size_t)(bm + cr) * K + cc;
  const _Float16* Bb0 = Bt + (size_t)(bn + cr) * K + cc;
  const _Float16* Bb1 = Bt + (size_t)(bn + cr + 64) * K + cc;

  int nk = K / 32;
  // preload tile 0
  async_cp16(&As[0][cr][cc],      Ab);
  async_cp16(&Bs[0][cr][cc],      Bb0);
  async_cp16(&Bs[0][cr + 64][cc], Bb1);

  for (int it = 0; it < nk; ++it) {
    wait_async0();
    __syncthreads();
    if (it + 1 < nk) {
      int p = (it + 1) & 1, k0 = (it + 1) * 32;
      async_cp16(&As[p][cr][cc],      Ab + k0);
      async_cp16(&Bs[p][cr][cc],      Bb0 + k0);
      async_cp16(&Bs[p][cr + 64][cc], Bb1 + k0);
    }
    int p = it & 1;
    v16h a0 = ld_frag(&As[p][wr * 32 + ln][0],      g);
    v16h a1 = ld_frag(&As[p][wr * 32 + 16 + ln][0], g);
    v16h b0 = ld_frag(&Bs[p][wc * 32 + ln][0],      g);
    v16h b1 = ld_frag(&Bs[p][wc * 32 + 16 + ln][0], g);
    acc[0] = WMMA_F32_F16(a0, b0, acc[0]);
    acc[1] = WMMA_F32_F16(a0, b1, acc[1]);
    acc[2] = WMMA_F32_F16(a1, b0, acc[2]);
    acc[3] = WMMA_F32_F16(a1, b1, acc[3]);
  }

#pragma unroll
  for (int mh = 0; mh < 2; ++mh)
#pragma unroll
    for (int nh = 0; nh < 2; ++nh) {
      v8f c = acc[mh * 2 + nh];
      int col = bn + wc * 32 + nh * 16 + ln;
      if (col < N) {                                 // only trims at V edge
        float bs = bias ? bias[col] : 0.0f;
#pragma unroll
        for (int i = 0; i < 8; ++i) {
          int row = bm + wr * 32 + mh * 16 + (lane >> 4) * 8 + i;
          float v = c[i] + bs;
          if (act == 1) v = (v >= 0.0f) ? v : 0.01f * v;  // leaky_relu(0.01)
          if (resid) v += resid[(size_t)row * N + col];
          if (outf) outf[(size_t)row * N + col] = v;
          if (outh) outh[(size_t)row * N + col] = (_Float16)v;
        }
      }
    }
}

// ---------------------------------------------------------------------------
// Causal attention with online softmax (flash style), WMMA for QK^T and PV.
// qkv layout: [B*S, 2304] f16, per-head chunks of 192: [q(64)|k(64)|v(64)].
// Grid: (S/64, H, B); block = 128 (4 waves, each owns 16 q-rows).
// ---------------------------------------------------------------------------
__global__ __launch_bounds__(128) void attn_kernel(const _Float16* __restrict__ qkv,
                                                   _Float16* __restrict__ ao) {
  __shared__ alignas(16) _Float16 Pl[4][16][40];
  int warp = threadIdx.x >> 5, lane = threadIdx.x & 31;
  int ln = lane & 15, half = lane >> 4, g = half * 8;
  int qb = blockIdx.x, h = blockIdx.y, b = blockIdx.z;
  int q0 = qb * 64 + warp * 16;
  size_t rb = (size_t)b * Sc;

  const _Float16* qbase = qkv + (rb + q0 + ln) * D3 + h * (3 * DHc);
  v16h qf[2];
  qf[0] = ld_frag(qbase, g);
  qf[1] = ld_frag(qbase + 32, g);

  v8f o[4];
  float Mr[8], Sr[8];
#pragma unroll
  for (int j = 0; j < 4; ++j)
#pragma unroll
    for (int i = 0; i < 8; ++i) o[j][i] = 0.0f;
#pragma unroll
  for (int i = 0; i < 8; ++i) { Mr[i] = -1e30f; Sr[i] = 0.0f; }

  int ntiles = 2 * qb + 2;
  for (int kt = 0; kt < ntiles; ++kt) {
    v8f s0, s1;
#pragma unroll
    for (int i = 0; i < 8; ++i) { s0[i] = 0.0f; s1[i] = 0.0f; }
    const _Float16* k0p = qkv + (rb + kt * 32 + ln) * D3 + h * (3 * DHc) + DHc;
    const _Float16* k1p = k0p + (size_t)16 * D3;
#pragma unroll
    for (int kh = 0; kh < 2; ++kh) {
      v16h kb0 = ld_frag(k0p + kh * 32, g);
      v16h kb1 = ld_frag(k1p + kh * 32, g);
      s0 = WMMA_F32_F16(qf[kh], kb0, s0);
      s1 = WMMA_F32_F16(qf[kh], kb1, s1);
    }
    int key0 = kt * 32 + ln, key1 = key0 + 16;
#pragma unroll
    for (int i = 0; i < 8; ++i) {
      int qi = q0 + i + half * 8;
      float a = s0[i] * 0.125f; if (key0 > qi) a = -100000.0f;
      float c = s1[i] * 0.125f; if (key1 > qi) c = -100000.0f;
      s0[i] = a; s1[i] = c;
    }
    float p0[8], p1[8], fsc[8];
#pragma unroll
    for (int i = 0; i < 8; ++i) {
      float v = fmaxf(s0[i], s1[i]);
#pragma unroll
      for (int off = 1; off < 16; off <<= 1) v = fmaxf(v, __shfl_xor(v, off));
      float Mn = fmaxf(Mr[i], v);
      float f  = __expf(Mr[i] - Mn);
      Mr[i] = Mn; fsc[i] = f;
      float a = __expf(s0[i] - Mn), c = __expf(s1[i] - Mn);
      float rs = a + c;
#pragma unroll
      for (int off = 1; off < 16; off <<= 1) rs += __shfl_xor(rs, off);
      Sr[i] = Sr[i] * f + rs;
      p0[i] = a; p1[i] = c;
    }
#pragma unroll
    for (int j = 0; j < 4; ++j)
#pragma unroll
      for (int i = 0; i < 8; ++i) o[j][i] *= fsc[i];

#pragma unroll
    for (int i = 0; i < 8; ++i) {
      Pl[warp][i + half * 8][ln]      = (_Float16)p0[i];
      Pl[warp][i + half * 8][16 + ln] = (_Float16)p1[i];
    }
    __syncthreads();
    v16h pa = ld_frag(&Pl[warp][ln][0], g);

    const _Float16* vbase = qkv + rb * D3 + h * (3 * DHc) + 2 * DHc + ln;
#pragma unroll
    for (int j = 0; j < 4; ++j) {
      v16h vb;
#pragma unroll
      for (int e = 0; e < 8; ++e) {
        int kk0 = g + e, kk1 = 16 + g + e;
        vb[e]     = vbase[(size_t)(kt * 32 + kk0) * D3 + j * 16];
        vb[e + 8] = vbase[(size_t)(kt * 32 + kk1) * D3 + j * 16];
      }
      o[j] = WMMA_F32_F16(pa, vb, o[j]);
    }
    __syncthreads();
  }

#pragma unroll
  for (int i = 0; i < 8; ++i) {
    float inv = 1.0f / Sr[i];
    int row = q0 + i + half * 8;
#pragma unroll
    for (int j = 0; j < 4; ++j) {
      ao[(rb + row) * Dc + h * DHc + j * 16 + ln] = (_Float16)(o[j][i] * inv);
    }
  }
}

// ---------------------------------------------------------------------------
__global__ void cvt_kernel(const float* __restrict__ in, _Float16* __restrict__ out, int n) {
  int i = blockIdx.x * blockDim.x + threadIdx.x;
  if (i < n) out[i] = (_Float16)in[i];
}

// ---------------------------------------------------------------------------
extern "C" void kernel_launch(void* const* d_in, const int* in_sizes, int n_in,
                              void* d_out, int out_size, void* d_ws, size_t ws_size,
                              hipStream_t stream) {
  (void)in_sizes; (void)n_in; (void)out_size; (void)ws_size;
  const int*   vocab = (const int*)d_in[0];
  const int*   pos   = (const int*)d_in[1];
  const float* vW    = (const float*)d_in[2];
  const float* pW    = (const float*)d_in[3];
  const float* ln1s  = (const float*)d_in[4];
  const float* ln1b  = (const float*)d_in[5];
  const float* Wqkv  = (const float*)d_in[6];
  const float* bqkv  = (const float*)d_in[7];
  const float* Wo    = (const float*)d_in[8];
  const float* bo    = (const float*)d_in[9];
  const float* ln2s  = (const float*)d_in[10];
  const float* ln2b  = (const float*)d_in[11];
  const float* W1    = (const float*)d_in[12];
  const float* b1    = (const float*)d_in[13];
  const float* W2    = (const float*)d_in[14];
  const float* b2    = (const float*)d_in[15];
  const float* outW  = (const float*)d_in[16];
  float* logits = (float*)d_out;

  char* p = (char*)d_ws;
  auto alloc = [&](size_t bytes) -> char* {
    char* r = p; p += (bytes + 255) & ~(size_t)255; return r;
  };
  float*    x_f   = (float*)   alloc((size_t)Rc * Dc * 4);
  _Float16* n1_h  = (_Float16*)alloc((size_t)Rc * Dc * 2);
  _Float16* qkv_h = (_Float16*)alloc((size_t)Rc * D3 * 2);
  _Float16* ao_h  = (_Float16*)alloc((size_t)Rc * Dc * 2);
  float*    att_f = (float*)   alloc((size_t)Rc * Dc * 4);
  float*    n2_f  = (float*)   alloc((size_t)Rc * Dc * 4);
  _Float16* n2_h  = (_Float16*)alloc((size_t)Rc * Dc * 2);
  _Float16* h_h   = (_Float16*)alloc((size_t)Rc * HIDc * 2);
  _Float16* x_h   = (_Float16*)alloc((size_t)Rc * Dc * 2);
  _Float16* wt    = (_Float16*)alloc((size_t)Vpad * Dc * 2);  // shared transposed-weight buf

  dim3 tb(32, 8);
  embed_kernel<<<(Rc * Dc + 255) / 256, 256, 0, stream>>>(vocab, pos, vW, pW, x_f);

  for (int l = 0; l < Lc; ++l) {
    ln_kernel<<<Rc, 256, 0, stream>>>(x_f, ln1s + l * Dc, ln1b + l * Dc, nullptr, n1_h);

    wt_kernel<<<dim3(D3 / 32, Dc / 32), tb, 0, stream>>>(Wqkv + (size_t)l * Dc * D3, wt, Dc, D3);
    gemm_kernel<<<dim3(D3 / 128, Rc / 64), 256, 0, stream>>>(
        n1_h, wt, bqkv + l * D3, nullptr, nullptr, qkv_h, D3, Dc, 0);

    attn_kernel<<<dim3(Sc / 64, Hc, Bc), 128, 0, stream>>>(qkv_h, ao_h);

    wt_kernel<<<dim3(Dc / 32, Dc / 32), tb, 0, stream>>>(Wo + (size_t)l * Dc * Dc, wt, Dc, Dc);
    gemm_kernel<<<dim3(Dc / 128, Rc / 64), 256, 0, stream>>>(
        ao_h, wt, bo + l * Dc, x_f, att_f, nullptr, Dc, Dc, 0);

    ln_kernel<<<Rc, 256, 0, stream>>>(att_f, ln2s + l * Dc, ln2b + l * Dc, n2_f, n2_h);

    wt_kernel<<<dim3(HIDc / 32, Dc / 32), tb, 0, stream>>>(W1 + (size_t)l * Dc * HIDc, wt, Dc, HIDc);
    gemm_kernel<<<dim3(HIDc / 128, Rc / 64), 256, 0, stream>>>(
        n2_h, wt, b1 + l * HIDc, nullptr, nullptr, h_h, HIDc, Dc, 1);

    wt_kernel<<<dim3(Dc / 32, HIDc / 32), tb, 0, stream>>>(W2 + (size_t)l * HIDc * Dc, wt, HIDc, Dc);
    gemm_kernel<<<dim3(Dc / 128, Rc / 64), 256, 0, stream>>>(
        h_h, wt, b2 + l * Dc, n2_f, x_f, nullptr, Dc, HIDc, 0);  // residual = n2 (faithful)
  }

  cvt_kernel<<<(Rc * Dc + 255) / 256, 256, 0, stream>>>(x_f, x_h, Rc * Dc);
  wt_kernel<<<dim3(Vpad / 32, Dc / 32), tb, 0, stream>>>(outW, wt, Dc, Vc);
  gemm_kernel<<<dim3(Vpad / 128, Rc / 64), 256, 0, stream>>>(
      x_h, wt, nullptr, nullptr, logits, nullptr, Vc, Dc, 0);
}
